// BaseAttention_64029372449516
// MI455X (gfx1250) — matrix operand order repaired and checked by reading, not compiled, not run on hardware
//
#include <hip/hip_runtime.h>

// ---------------------------------------------------------------------------
// Multi-head attention forward for MI455X (gfx1250), wave32 + WMMA +
// async global->LDS staging (ASYNCcnt) with double buffering.
// B=2, T=2048, D=2048, H=16, HD=128.
// ---------------------------------------------------------------------------

#define BB 2
#define TT 2048
#define DD 2048
#define HH 16
#define HD 128
#define ATTN_SCALE 0.08838834764831845f  // 1/sqrt(128)

typedef __attribute__((ext_vector_type(16))) _Float16 v16h;
typedef __attribute__((ext_vector_type(8)))  _Float16 v8h;
typedef __attribute__((ext_vector_type(8)))  float    v8f;
typedef int v4i __attribute__((vector_size(4 * sizeof(int))));

static __device__ __forceinline__ v16h cat8(v8h lo, v8h hi) {
  return __builtin_shufflevector(lo, hi, 0,1,2,3,4,5,6,7,8,9,10,11,12,13,14,15);
}

// Async global->LDS copy (CDNA5 GLOBAL_LOAD_ASYNC_TO_LDS_B128, ASYNCcnt).
#if defined(__gfx1250__) && \
    __has_builtin(__builtin_amdgcn_global_load_async_to_lds_b128) && \
    __has_builtin(__builtin_amdgcn_s_wait_asynccnt)
#define USE_ASYNC_LDS 1
#else
#define USE_ASYNC_LDS 0
#endif

#if USE_ASYNC_LDS
#define LDS_AS __attribute__((address_space(3)))
#define GLB_AS __attribute__((address_space(1)))
static __device__ __forceinline__ void async_b128(const _Float16* gsrc,
                                                  _Float16* ldst) {
  __builtin_amdgcn_global_load_async_to_lds_b128(
      (GLB_AS v4i*)gsrc, (LDS_AS v4i*)ldst, 0, 0);
}
#endif

// ---------------------------------------------------------------------------
// fp32 -> fp16 conversion
// ---------------------------------------------------------------------------
__global__ void cvt_f32_to_f16(const float* __restrict__ src,
                               _Float16* __restrict__ dst, int n) {
  int i = blockIdx.x * blockDim.x + threadIdx.x;
  if (i < n) dst[i] = (_Float16)src[i];
}

// ---------------------------------------------------------------------------
// Y[M,N] = A[M,K] @ W[N,K]^T + bias[N]
// Block tile 128(M) x 256(N), K-step 32. 256 threads = 8 waves (2 M x 4 N),
// each wave computes 64x64 via 4x4 accumulators of v_wmma_f32_16x16x32_f16.
// Double-buffered LDS fed by async global->LDS loads: one barrier per K-step,
// next tile's loads overlap current tile's WMMAs.
// ---------------------------------------------------------------------------
template <typename OutT>
__global__ __launch_bounds__(256)
void gemm_bias_kernel(const _Float16* __restrict__ A,
                      const _Float16* __restrict__ W,
                      const float* __restrict__ bias,
                      OutT* __restrict__ Y,
                      int M, int N, int K) {
  __shared__ __align__(32) _Float16 As[2][128 * 32];  // 16 KB
  __shared__ __align__(32) _Float16 Bs[2][256 * 32];  // 32 KB

  const int t    = threadIdx.x;
  const int lane = t & 31;
  const int wave = t >> 5;
  const int wm   = wave >> 2;   // 0..1  (M direction, 64 rows each)
  const int wn   = wave & 3;    // 0..3  (N direction, 64 cols each)
  const int hh   = lane >> 4;
  const int lr   = lane & 15;

  const int m0 = blockIdx.x * 128;
  const int n0 = blockIdx.y * 256;
  const int NK = K >> 5;

  const v8f vzero = {};
  v8f acc[4][4];
#pragma unroll
  for (int mi = 0; mi < 4; ++mi)
#pragma unroll
    for (int ni = 0; ni < 4; ++ni) acc[mi][ni] = vzero;

#if USE_ASYNC_LDS
  auto issue = [&](int kt, int buf) {
    const int k0 = kt << 5;
#pragma unroll
    for (int i = 0; i < 2; ++i) {       // A tile: 512 uint4
      int idx = t + i * 256, row = idx >> 2, c = idx & 3;
      async_b128(A + (size_t)(m0 + row) * K + k0 + c * 8, &As[buf][idx * 8]);
    }
#pragma unroll
    for (int i = 0; i < 4; ++i) {       // B tile: 1024 uint4
      int idx = t + i * 256, row = idx >> 2, c = idx & 3;
      async_b128(W + (size_t)(n0 + row) * K + k0 + c * 8, &Bs[buf][idx * 8]);
    }
  };
  issue(0, 0);
#endif

  for (int kt = 0; kt < NK; ++kt) {
#if USE_ASYNC_LDS
    __builtin_amdgcn_s_wait_asynccnt(0);  // tile kt landed in LDS
    __syncthreads();                      // all waves landed; prev reads done
    if (kt + 1 < NK) issue(kt + 1, (kt + 1) & 1);
    const _Float16* as = As[kt & 1];
    const _Float16* bs = Bs[kt & 1];
#else
    const int k0 = kt << 5;
    __syncthreads();
#pragma unroll
    for (int i = 0; i < 2; ++i) {
      int idx = t + i * 256, row = idx >> 2, c = idx & 3;
      *(uint4*)&As[0][idx * 8] =
          *(const uint4*)(A + (size_t)(m0 + row) * K + k0 + c * 8);
    }
#pragma unroll
    for (int i = 0; i < 4; ++i) {
      int idx = t + i * 256, row = idx >> 2, c = idx & 3;
      *(uint4*)&Bs[0][idx * 8] =
          *(const uint4*)(W + (size_t)(n0 + row) * K + k0 + c * 8);
    }
    __syncthreads();
    const _Float16* as = As[0];
    const _Float16* bs = Bs[0];
#endif

    // A fragments (16x32 f16 layout): lane lr = M row, halves split K.
    v16h af[4];
#pragma unroll
    for (int mi = 0; mi < 4; ++mi) {
      const _Float16* r = as + (wm * 64 + mi * 16 + lr) * 32;
      af[mi] = cat8(*(const v8h*)(r + hh * 8), *(const v8h*)(r + 16 + hh * 8));
    }
    // B fragments (32x16): lane lr = N col, same K split as A fragments.
#pragma unroll
    for (int ni = 0; ni < 4; ++ni) {
      const _Float16* r = bs + (wn * 64 + ni * 16 + lr) * 32;
      v16h bf = cat8(*(const v8h*)(r + hh * 8), *(const v8h*)(r + 16 + hh * 8));
#pragma unroll
      for (int mi = 0; mi < 4; ++mi) {
        acc[mi][ni] = __builtin_amdgcn_wmma_f32_16x16x32_f16(
            false, af[mi], false, bf, (short)0, acc[mi][ni], false, false);
      }
    }
  }

  // Epilogue: C/D layout (M = r + 8*hh, N = lr per tile) -> global + bias.
#pragma unroll
  for (int mi = 0; mi < 4; ++mi) {
#pragma unroll
    for (int ni = 0; ni < 4; ++ni) {
      int col  = n0 + wn * 64 + ni * 16 + lr;
      float bv = bias[col];
#pragma unroll
      for (int r = 0; r < 8; ++r) {
        int row = m0 + wm * 64 + mi * 16 + r + 8 * hh;
        Y[(size_t)row * N + col] = (OutT)(acc[mi][ni][r] + bv);
      }
    }
  }
}

// ---------------------------------------------------------------------------
// Flash attention: grid (B*H, T/128). 256 threads = 8 waves; each wave owns
// 16 query rows. Keys streamed 64 at a time with online softmax.
// K tile staged via async global->LDS; V tile staged transposed manually.
// ---------------------------------------------------------------------------
__global__ __launch_bounds__(256)
void flash_attn_kernel(const _Float16* __restrict__ Qb,
                       const _Float16* __restrict__ Kb,
                       const _Float16* __restrict__ Vb,
                       _Float16* __restrict__ Ctx) {
  __shared__ __align__(32) _Float16 Ks[64 * 128];   // K tile [key][hd]
  __shared__ __align__(32) _Float16 Vt[128 * 64];   // V^T tile [hd][key]
  __shared__ __align__(32) _Float16 Ps[8][16 * 64]; // per-wave P patch

  const int bh = blockIdx.x;
  const int b  = bh >> 4;
  const int h  = bh & 15;
  const int q0 = blockIdx.y * 128;

  const int t    = threadIdx.x;
  const int lane = t & 31;
  const int wave = t >> 5;
  const int hh   = lane >> 4;
  const int lr   = lane & 15;

  const size_t base = (size_t)b * TT * DD + (size_t)h * HD;

  // Q fragments for this wave's 16 rows (persist across the key loop).
  const int qrow = q0 + wave * 16 + lr;
  v16h qf[4];
#pragma unroll
  for (int f = 0; f < 4; ++f) {
    const _Float16* r = Qb + base + (size_t)qrow * DD + f * 32;
    qf[f] = cat8(*(const v8h*)(r + hh * 8), *(const v8h*)(r + 16 + hh * 8));
  }

  const v8f vzero = {};
  v8f o[8];
#pragma unroll
  for (int ht = 0; ht < 8; ++ht) o[ht] = vzero;

  float mrow[8], lrow[8];
#pragma unroll
  for (int r = 0; r < 8; ++r) { mrow[r] = -__builtin_inff(); lrow[r] = 0.0f; }

  const int jend = q0 + 128;
  for (int j0 = 0; j0 < jend; j0 += 64) {
    __syncthreads();  // previous-iteration reads of Ks/Vt are done

    // Stage K tile (async copy) and V tile (transposed, via registers).
#pragma unroll
    for (int i = 0; i < 4; ++i) {
      int idx = t + i * 256;          // 0..1023 over 64x128 halfs
      int row = idx >> 3;
      int c   = idx & 7;
#if USE_ASYNC_LDS
      async_b128(Kb + base + (size_t)(j0 + row) * DD + c * 8,
                 Ks + row * 128 + c * 8);
#else
      *(uint4*)(Ks + row * 128 + c * 8) =
          *(const uint4*)(Kb + base + (size_t)(j0 + row) * DD + c * 8);
#endif
      union { uint4 u; _Float16 hv[8]; } vv;
      vv.u = *(const uint4*)(Vb + base + (size_t)(j0 + row) * DD + c * 8);
#pragma unroll
      for (int e = 0; e < 8; ++e) Vt[(c * 8 + e) * 64 + row] = vv.hv[e];
      // Prefetch next key chunk into cache while we compute this one.
      if (j0 + 64 < jend) {
        __builtin_prefetch(Kb + base + (size_t)(j0 + 64 + row) * DD + c * 8, 0, 1);
        __builtin_prefetch(Vb + base + (size_t)(j0 + 64 + row) * DD + c * 8, 0, 1);
      }
    }
#if USE_ASYNC_LDS
    __builtin_amdgcn_s_wait_asynccnt(0);
#endif
    __syncthreads();

    // S = Q K^T : 4 key tiles x 4 K-dim steps of WMMA.
    v8f s[4];
#pragma unroll
    for (int kt = 0; kt < 4; ++kt) {
      v8f sv = vzero;
#pragma unroll
      for (int f = 0; f < 4; ++f) {
        const _Float16* kr = Ks + (kt * 16 + lr) * 128 + f * 32;
        v16h kfv = cat8(*(const v8h*)(kr + hh * 8), *(const v8h*)(kr + 16 + hh * 8));
        sv = __builtin_amdgcn_wmma_f32_16x16x32_f16(
            false, qf[f], false, kfv, (short)0, sv, false, false);
      }
      s[kt] = sv;
    }

    // Scale + causal mask (C layout: M = r + 8*hh, N = lr).
#pragma unroll
    for (int kt = 0; kt < 4; ++kt) {
      int key = j0 + kt * 16 + lr;
#pragma unroll
      for (int r = 0; r < 8; ++r) {
        int q   = q0 + wave * 16 + r + 8 * hh;
        float v = s[kt][r] * ATTN_SCALE;
        s[kt][r] = (key > q) ? -__builtin_inff() : v;
      }
    }

    // Online softmax across the 64 keys (row reductions within lane half).
    float alpha[8];
#pragma unroll
    for (int r = 0; r < 8; ++r) {
      float mx = s[0][r];
#pragma unroll
      for (int kt = 1; kt < 4; ++kt) mx = fmaxf(mx, s[kt][r]);
      for (int off = 1; off < 16; off <<= 1)
        mx = fmaxf(mx, __shfl_xor(mx, off, 32));
      float mnew = fmaxf(mrow[r], mx);
      alpha[r]   = __expf(mrow[r] - mnew);
      mrow[r]    = mnew;
      float sum = 0.0f;
#pragma unroll
      for (int kt = 0; kt < 4; ++kt) {
        float p = __expf(s[kt][r] - mnew);
        s[kt][r] = p;
        sum += p;
      }
      for (int off = 1; off < 16; off <<= 1)
        sum += __shfl_xor(sum, off, 32);
      lrow[r] = lrow[r] * alpha[r] + sum;
    }
#pragma unroll
    for (int ht = 0; ht < 8; ++ht)
#pragma unroll
      for (int r = 0; r < 8; ++r) o[ht][r] *= alpha[r];

    // P: C layout -> A layout via per-wave LDS patch (wave-local, ds in-order).
    _Float16* pw = Ps[wave];
#pragma unroll
    for (int kt = 0; kt < 4; ++kt)
#pragma unroll
      for (int r = 0; r < 8; ++r)
        pw[(r + 8 * hh) * 64 + kt * 16 + lr] = (_Float16)s[kt][r];

    v16h pf[2];
#pragma unroll
    for (int kf = 0; kf < 2; ++kf) {
      const _Float16* rr = pw + lr * 64 + kf * 32;
      pf[kf] = cat8(*(const v8h*)(rr + hh * 8), *(const v8h*)(rr + 16 + hh * 8));
    }

    // O += P V : 8 hd tiles x 2 key-dim steps.
#pragma unroll
    for (int ht = 0; ht < 8; ++ht) {
#pragma unroll
      for (int kf = 0; kf < 2; ++kf) {
        const _Float16* vr = Vt + (ht * 16 + lr) * 64 + kf * 32;
        v16h vf = cat8(*(const v8h*)(vr + hh * 8), *(const v8h*)(vr + 16 + hh * 8));
        o[ht] = __builtin_amdgcn_wmma_f32_16x16x32_f16(
            false, pf[kf], false, vf, (short)0, o[ht], false, false);
      }
    }
  }

  // Normalize and store context (f16, [B*T, D] layout).
#pragma unroll
  for (int r = 0; r < 8; ++r) {
    float inv = 1.0f / lrow[r];
    int q = q0 + wave * 16 + r + 8 * hh;
#pragma unroll
    for (int ht = 0; ht < 8; ++ht) {
      Ctx[base + (size_t)q * DD + ht * 16 + lr] = (_Float16)(o[ht][r] * inv);
    }
  }
}

// ---------------------------------------------------------------------------
// Host-side orchestration.
// ---------------------------------------------------------------------------
extern "C" void kernel_launch(void* const* d_in, const int* in_sizes, int n_in,
                              void* d_out, int out_size, void* d_ws, size_t ws_size,
                              hipStream_t stream) {
  const float* x  = (const float*)d_in[0];
  // d_in[1] = mask (unused: causal mask computed analytically)
  const float* wq = (const float*)d_in[2];
  const float* bq = (const float*)d_in[3];
  const float* wk = (const float*)d_in[4];
  const float* bk = (const float*)d_in[5];
  const float* wv = (const float*)d_in[6];
  const float* bv = (const float*)d_in[7];
  const float* wo = (const float*)d_in[8];
  const float* bo = (const float*)d_in[9];
  float* out = (float*)d_out;

  const size_t NTD = (size_t)BB * TT * DD;  // 8388608
  const size_t NW  = (size_t)DD * DD;       // 4194304

  _Float16* p   = (_Float16*)d_ws;
  _Float16* xh  = p; p += NTD;
  _Float16* wqh = p; p += NW;
  _Float16* wkh = p; p += NW;
  _Float16* wvh = p; p += NW;
  _Float16* woh = p; p += NW;
  _Float16* qh  = p; p += NTD;
  _Float16* kh  = p; p += NTD;
  _Float16* vh  = p; p += NTD;
  _Float16* ch  = p; p += NTD;

  // 1) fp32 -> fp16 conversions.
  {
    int n = (int)NTD;
    cvt_f32_to_f16<<<(n + 255) / 256, 256, 0, stream>>>(x, xh, n);
    int m = (int)NW;
    int blk = (m + 255) / 256;
    cvt_f32_to_f16<<<blk, 256, 0, stream>>>(wq, wqh, m);
    cvt_f32_to_f16<<<blk, 256, 0, stream>>>(wk, wkh, m);
    cvt_f32_to_f16<<<blk, 256, 0, stream>>>(wv, wvh, m);
    cvt_f32_to_f16<<<blk, 256, 0, stream>>>(wo, woh, m);
  }

  const int M = BB * TT;          // 4096
  dim3 ggrid(M / 128, DD / 256);  // 32 x 8

  // 2) Q/K/V projections (f16 out).
  gemm_bias_kernel<_Float16><<<ggrid, 256, 0, stream>>>(xh, wqh, bq, qh, M, DD, DD);
  gemm_bias_kernel<_Float16><<<ggrid, 256, 0, stream>>>(xh, wkh, bk, kh, M, DD, DD);
  gemm_bias_kernel<_Float16><<<ggrid, 256, 0, stream>>>(xh, wvh, bv, vh, M, DD, DD);

  // 3) Flash attention.
  dim3 agrid(BB * HH, TT / 128);  // 32 x 16
  flash_attn_kernel<<<agrid, 256, 0, stream>>>(qh, kh, vh, ch);

  // 4) Output projection (f32 out, bias bo) straight into d_out.
  gemm_bias_kernel<float><<<ggrid, 256, 0, stream>>>(ch, woh, bo, out, M, DD, DD);
}
